// SwinTransformerBlock_20993800143081
// MI455X (gfx1250) — matrix operand order repaired and checked by reading, not compiled, not run on hardware
//
#include <hip/hip_runtime.h>
#include <hip/hip_bf16.h>

typedef __attribute__((ext_vector_type(16))) _Float16 v16h;
typedef __attribute__((ext_vector_type(8)))  float    v8f;

#define S_DIM 42
#define C_DIM 96
#define NWIN  216
#define NTOK  343     // 7^3 tokens / window
#define NPAD  352     // padded to 22*16
#define NHEAD 3
#define HD    32
#define T_TOK 74088   // 42^3
#define TW    (NWIN * NPAD)   // 76032 windowed (padded) rows; 4752 M-tiles
#define TP    74112           // tokens padded to 16*4632 (even tile count)
#define SHIFT3 3
#define QS    (NWIN * NHEAD * NPAD * HD)   // 7299072 elems per q/k/v plane
#define SCALE_Q 0.17677669529663687f       // 32^-0.5

// ---------------- WMMA fragment loaders (ISA 7.12.2 layouts, wave32) ----------

__device__ __forceinline__ v16h load_a_f16(const _Float16* p, int ld) {
  const int lane = threadIdx.x & 31;
  const int row  = lane & 15;
  const int kb   = (lane < 16) ? 0 : 8;
  v16h a;
#pragma unroll
  for (int i = 0; i < 8; ++i) a[i]     = p[row * ld + kb + i];
#pragma unroll
  for (int i = 0; i < 8; ++i) a[8 + i] = p[row * ld + 16 + kb + i];
  return a;
}

__device__ __forceinline__ v16h load_a_f32lds(const float* p, int ld) {
  const int lane = threadIdx.x & 31;
  const int row  = lane & 15;
  const int kb   = (lane < 16) ? 0 : 8;
  v16h a;
#pragma unroll
  for (int i = 0; i < 8; ++i) a[i]     = (_Float16)p[row * ld + kb + i];
#pragma unroll
  for (int i = 0; i < 8; ++i) a[8 + i] = (_Float16)p[row * ld + 16 + kb + i];
  return a;
}

// B fragment where memory holds B^T row-major (mem[n][k], ld = K stride):
// per-lane two 16B chunks (k = kb..kb+7, 16+kb..16+kb+7) -> 2x global_load_b128.
__device__ __forceinline__ v16h load_bT_f16(const _Float16* p, int ld) {
  const int lane = threadIdx.x & 31;
  const int col  = lane & 15;
  const int kb   = (lane < 16) ? 0 : 8;
  v16h b;
#pragma unroll
  for (int i = 0; i < 8; ++i) b[i]     = p[col * ld + kb + i];
#pragma unroll
  for (int i = 0; i < 8; ++i) b[8 + i] = p[col * ld + 16 + kb + i];
  return b;
}

__device__ __forceinline__ v8f wmma_f16(v16h a, v16h b, v8f c) {
  return __builtin_amdgcn_wmma_f32_16x16x32_f16(false, a, false, b, (short)0, c,
                                                false, false);
}

// ---------------- weight f32 [K][N] -> f16 transposed [N][K] ------------------
// grid: (ceil(N/256), K) -> no div/mod

__global__ void k_cvt_t(const float* __restrict__ s, _Float16* __restrict__ d,
                        int K, int N) {
  const int n = blockIdx.x * 256 + threadIdx.x;
  const int k = blockIdx.y;
  if (n < N) d[(size_t)n * K + k] = (_Float16)s[(size_t)k * N + n];
}

// ---------------- LN1 + cyclic shift + window partition (f16 out) ------------
// grid: (44 n-blocks, 216 windows); one wave per windowed row

__global__ void k_ln1_window(const float* __restrict__ x,
                             const float* __restrict__ w, const float* __restrict__ b,
                             _Float16* __restrict__ hw) {
  const int wave = threadIdx.x >> 5, lane = threadIdx.x & 31;
  const int win = blockIdx.y;
  const int n = blockIdx.x * 8 + wave;           // 0..351
  _Float16* op = hw + ((size_t)win * NPAD + n) * C_DIM;
  if (n >= NTOK) {
    op[lane] = (_Float16)0.f; op[lane + 32] = (_Float16)0.f; op[lane + 64] = (_Float16)0.f;
    return;
  }
  const int wz = win / 36, wy = (win / 6) % 6, wx = win % 6;
  const int z = n / 49, y = (n / 7) % 7, xx = n % 7;
  const int ss = (wz * 7 + z  + SHIFT3) % S_DIM;
  const int hs = (wy * 7 + y  + SHIFT3) % S_DIM;
  const int ws = (wx * 7 + xx + SHIFT3) % S_DIM;
  const int t  = (ss * S_DIM + hs) * S_DIM + ws;
  const float* xp = x + (size_t)t * C_DIM;
  float v0 = xp[lane], v1 = xp[lane + 32], v2 = xp[lane + 64];
  float s = v0 + v1 + v2, sq = v0 * v0 + v1 * v1 + v2 * v2;
#pragma unroll
  for (int o = 16; o; o >>= 1) { s += __shfl_xor(s, o, 32); sq += __shfl_xor(sq, o, 32); }
  const float mu = s * (1.f / 96.f);
  const float rs = rsqrtf(sq * (1.f / 96.f) - mu * mu + 1e-5f);
  op[lane]      = (_Float16)((v0 - mu) * rs * w[lane]      + b[lane]);
  op[lane + 32] = (_Float16)((v1 - mu) * rs * w[lane + 32] + b[lane + 32]);
  op[lane + 64] = (_Float16)((v2 - mu) * rs * w[lane + 64] + b[lane + 64]);
}

// ---------------- QKV GEMM: 2x2 micro-tile; V stored transposed --------------
// grid: (297, 9): wave = M-tile pair, blockIdx.y = N-tile pair

__global__ void k_qkv(const _Float16* __restrict__ hw, const _Float16* __restrict__ wqT,
                      const float* __restrict__ bq, _Float16* __restrict__ qkv) {
  const int wave = threadIdx.x >> 5, lane = threadIdx.x & 31;
  const int mt0 = (blockIdx.x * 8 + wave) * 2, mt1 = mt0 + 1;
  const int nt0 = blockIdx.y * 2;
  __builtin_prefetch(wqT + (size_t)nt0 * 16 * C_DIM, 0, 1);
  v8f a00 = {}, a01 = {}, a10 = {}, a11 = {};
#pragma unroll
  for (int kt = 0; kt < 3; ++kt) {
    v16h A0 = load_a_f16(hw + (size_t)mt0 * 16 * C_DIM + kt * 32, C_DIM);
    v16h A1 = load_a_f16(hw + (size_t)mt1 * 16 * C_DIM + kt * 32, C_DIM);
    v16h B0 = load_bT_f16(wqT + ((size_t)nt0 * 16) * C_DIM + kt * 32, C_DIM);
    v16h B1 = load_bT_f16(wqT + ((size_t)(nt0 + 1) * 16) * C_DIM + kt * 32, C_DIM);
    a00 = wmma_f16(A0, B0, a00); a01 = wmma_f16(A0, B1, a01);
    a10 = wmma_f16(A1, B0, a10); a11 = wmma_f16(A1, B1, a11);
  }
  const int col = lane & 15, rbase = (lane < 16) ? 0 : 8;
#pragma unroll
  for (int mh = 0; mh < 2; ++mh) {
    const int mt = mh ? mt1 : mt0;
    const int win = mt / 22, nb = (mt % 22) * 16;   // tiles never cross windows
#pragma unroll
    for (int nh = 0; nh < 2; ++nh) {
      const v8f acc = mh ? (nh ? a11 : a10) : (nh ? a01 : a00);
      const int cg = (nt0 + nh) * 16 + col;
      const int which = cg / 96, r = cg % 96, head = r / 32, hd = r % 32;
      const float bias = bq[cg];
      const float sc = (which == 0) ? SCALE_Q : 1.f;
      const size_t plane = (size_t)(win * NHEAD + head);
#pragma unroll
      for (int j = 0; j < 8; ++j) {
        const int n = nb + rbase + j;
        const float v = (acc[j] + bias) * sc;
        size_t idx;
        if (which == 2)  // V^T: [plane][hd][token]
          idx = 2 * (size_t)QS + (plane * HD + hd) * NPAD + n;
        else             // Q, K: [plane][token][hd]
          idx = (size_t)which * QS + (plane * NPAD + n) * HD + hd;
        qkv[idx] = (_Float16)v;
      }
    }
  }
}

// ---------------- attention: QK^T + bias + mask + softmax + PV ---------------
// grid: (648 win*head, 3 qblocks); 8 waves; wave owns 16 Q rows; 176 KB LDS

__global__ void k_attn(const _Float16* __restrict__ qkv,
                       const float* __restrict__ rpb, const float* __restrict__ maskm,
                       _Float16* __restrict__ ao) {
  extern __shared__ float smem[];
  const int wave = threadIdx.x >> 5, lane = threadIdx.x & 31;
  const int wh = blockIdx.x;                 // win*3 + head
  const int rt = blockIdx.y * 8 + wave;      // row tile 0..21
  if (rt >= NPAD / 16) return;
  const int win = wh / NHEAD, hh = wh - win * NHEAD;
  float* myl = smem + (size_t)wave * 16 * NPAD;
  const size_t base = (size_t)wh * NPAD * HD;
  const _Float16* qp = qkv + base;
  const _Float16* kp = qkv + (size_t)QS + base;
  const _Float16* vp = qkv + 2 * (size_t)QS + base;   // V^T [hd][token]
  const int col = lane & 15, rbase = (lane < 16) ? 0 : 8;

  const v16h qa = load_a_f16(qp + rt * 16 * HD, HD);  // full head dim (K=32)

  // hoist per-row decomposition (8 rows this lane writes)
  int nrow[8], zy20[8];  // zy20 = z*20 + precomputed? keep z,y,x separately
  int zn[8], yn[8], xn[8];
#pragma unroll
  for (int j = 0; j < 8; ++j) {
    const int n = rt * 16 + rbase + j;
    nrow[j] = n;
    zn[j] = n / 49; yn[j] = (n / 7) % 7; xn[j] = n % 7;
    zy20[j] = 0;
  }
  (void)zy20;

  // scores -> LDS, fused relative-position bias + window mask
  for (int ct = 0; ct < NPAD / 16; ++ct) {
    v16h kb = load_bT_f16(kp + ct * 16 * HD, HD);
    v8f acc = {};
    acc = wmma_f16(qa, kb, acc);
    const int m = ct * 16 + col;
    const int z2 = m / 49, y2 = (m / 7) % 7, x2 = m % 7;   // once per tile
    const int mvalid = (m < NTOK);
#pragma unroll
    for (int j = 0; j < 8; ++j) {
      const int nl = rbase + j;
      const int n = nrow[j];
      float v = acc[j];
      if (mvalid && n < NTOK) {
        const int idx = (zn[j] - z2 + 6) * 20 + (yn[j] - y2 + 6) * 13 + (xn[j] - x2 + 6);
        v += rpb[idx * NHEAD + hh] + maskm[((size_t)win * NTOK + n) * NTOK + m];
      }
      myl[nl * NPAD + m] = v;
    }
  }

  // row softmax (padded cols -> 0 probability)
  for (int r = 0; r < 16; ++r) {
    float* rp = myl + r * NPAD;
    float mx = -3.0e38f;
    for (int c = lane; c < NTOK; c += 32) mx = fmaxf(mx, rp[c]);
#pragma unroll
    for (int o = 16; o; o >>= 1) mx = fmaxf(mx, __shfl_xor(mx, o, 32));
    float sum = 0.f;
    for (int c = lane; c < NPAD; c += 32) {
      const float e = (c < NTOK) ? __expf(rp[c] - mx) : 0.f;
      rp[c] = e; sum += e;
    }
#pragma unroll
    for (int o = 16; o; o >>= 1) sum += __shfl_xor(sum, o, 32);
    const float inv = 1.f / sum;
    for (int c = lane; c < NPAD; c += 32) rp[c] *= inv;
  }

  // P @ V  (K = 352 = 11 x 32); two N-tiles with independent accumulators
  v8f o0 = {}, o1 = {};
#pragma unroll
  for (int kt = 0; kt < 11; ++kt) {
    v16h pa = load_a_f32lds(myl + kt * 32, NPAD);
    v16h v0 = load_bT_f16(vp + (size_t)0 * 16 * NPAD + kt * 32, NPAD);
    v16h v1 = load_bT_f16(vp + (size_t)1 * 16 * NPAD + kt * 32, NPAD);
    o0 = wmma_f16(pa, v0, o0);
    o1 = wmma_f16(pa, v1, o1);
  }
#pragma unroll
  for (int nt = 0; nt < 2; ++nt) {
    const v8f acc = nt ? o1 : o0;
    const int c = hh * HD + nt * 16 + col;
#pragma unroll
    for (int j = 0; j < 8; ++j)
      ao[((size_t)win * NPAD + nrow[j]) * C_DIM + c] = (_Float16)acc[j];
  }
}

// ---------------- proj GEMM + window reverse + unshift + residual ------------
// grid: (297, 3); 2x2 micro-tile; scatter indices precomputed per M-tile

__global__ void k_proj(const _Float16* __restrict__ ao, const _Float16* __restrict__ pwT,
                       const float* __restrict__ pb, const float* __restrict__ x,
                       float* __restrict__ xres) {
  const int wave = threadIdx.x >> 5, lane = threadIdx.x & 31;
  const int mt0 = (blockIdx.x * 8 + wave) * 2, mt1 = mt0 + 1;
  const int nt0 = blockIdx.y * 2;
  __builtin_prefetch(pwT, 0, 1);
  v8f a00 = {}, a01 = {}, a10 = {}, a11 = {};
#pragma unroll
  for (int kt = 0; kt < 3; ++kt) {
    v16h A0 = load_a_f16(ao + (size_t)mt0 * 16 * C_DIM + kt * 32, C_DIM);
    v16h A1 = load_a_f16(ao + (size_t)mt1 * 16 * C_DIM + kt * 32, C_DIM);
    v16h B0 = load_bT_f16(pwT + ((size_t)nt0 * 16) * C_DIM + kt * 32, C_DIM);
    v16h B1 = load_bT_f16(pwT + ((size_t)(nt0 + 1) * 16) * C_DIM + kt * 32, C_DIM);
    a00 = wmma_f16(A0, B0, a00); a01 = wmma_f16(A0, B1, a01);
    a10 = wmma_f16(A1, B0, a10); a11 = wmma_f16(A1, B1, a11);
  }
  const int col = lane & 15, rbase = (lane < 16) ? 0 : 8;
#pragma unroll
  for (int mh = 0; mh < 2; ++mh) {
    const int mt = mh ? mt1 : mt0;
    const int win = mt / 22, nb = (mt % 22) * 16;
    const int wz = win / 36, wy = (win / 6) % 6, wx = win % 6;
    // precompute scatter token index for this lane's 8 rows (once per M-tile)
    int tok[8];
#pragma unroll
    for (int j = 0; j < 8; ++j) {
      const int n = nb + rbase + j;
      if (n < NTOK) {
        const int z = n / 49, y = (n / 7) % 7, xx = n % 7;
        const int sf = (wz * 7 + z  + SHIFT3) % S_DIM;
        const int hf = (wy * 7 + y  + SHIFT3) % S_DIM;
        const int wf = (wx * 7 + xx + SHIFT3) % S_DIM;
        tok[j] = (sf * S_DIM + hf) * S_DIM + wf;
      } else tok[j] = -1;
    }
#pragma unroll
    for (int nh = 0; nh < 2; ++nh) {
      const v8f acc = mh ? (nh ? a11 : a10) : (nh ? a01 : a00);
      const int c = (nt0 + nh) * 16 + col;
      const float bias = pb[c];
#pragma unroll
      for (int j = 0; j < 8; ++j) {
        if (tok[j] >= 0) {
          const size_t d = (size_t)tok[j] * C_DIM + c;
          xres[d] = x[d] + acc[j] + bias;
        }
      }
    }
  }
}

// ---------------- LN2 (f16 out, rows padded to TP) ---------------------------

__global__ void k_ln2(const float* __restrict__ xr,
                      const float* __restrict__ w, const float* __restrict__ b,
                      _Float16* __restrict__ m16) {
  const int wave = threadIdx.x >> 5, lane = threadIdx.x & 31;
  const int t = blockIdx.x * 8 + wave;
  if (t >= TP) return;
  _Float16* op = m16 + (size_t)t * C_DIM;
  if (t >= T_TOK) {
    op[lane] = (_Float16)0.f; op[lane + 32] = (_Float16)0.f; op[lane + 64] = (_Float16)0.f;
    return;
  }
  const float* xp = xr + (size_t)t * C_DIM;
  float v0 = xp[lane], v1 = xp[lane + 32], v2 = xp[lane + 64];
  float s = v0 + v1 + v2, sq = v0 * v0 + v1 * v1 + v2 * v2;
#pragma unroll
  for (int o = 16; o; o >>= 1) { s += __shfl_xor(s, o, 32); sq += __shfl_xor(sq, o, 32); }
  const float mu = s * (1.f / 96.f);
  const float rs = rsqrtf(sq * (1.f / 96.f) - mu * mu + 1e-5f);
  op[lane]      = (_Float16)((v0 - mu) * rs * w[lane]      + b[lane]);
  op[lane + 32] = (_Float16)((v1 - mu) * rs * w[lane + 32] + b[lane + 32]);
  op[lane + 64] = (_Float16)((v2 - mu) * rs * w[lane + 64] + b[lane + 64]);
}

// ---------------- fc1 GEMM + exact (erf) GELU --------------------------------
// grid: (290, 12); 2x2 micro-tile

__global__ void k_fc1(const _Float16* __restrict__ m16, const _Float16* __restrict__ w1T,
                      const float* __restrict__ b1, _Float16* __restrict__ h1) {
  const int wave = threadIdx.x >> 5, lane = threadIdx.x & 31;
  const int mtp = blockIdx.x * 8 + wave;
  if (mtp >= (TP / 16) / 2) return;
  const int mt0 = mtp * 2, mt1 = mt0 + 1;
  const int nt0 = blockIdx.y * 2;
  __builtin_prefetch(w1T + (size_t)nt0 * 16 * C_DIM, 0, 1);
  v8f a00 = {}, a01 = {}, a10 = {}, a11 = {};
#pragma unroll
  for (int kt = 0; kt < 3; ++kt) {
    v16h A0 = load_a_f16(m16 + (size_t)mt0 * 16 * C_DIM + kt * 32, C_DIM);
    v16h A1 = load_a_f16(m16 + (size_t)mt1 * 16 * C_DIM + kt * 32, C_DIM);
    v16h B0 = load_bT_f16(w1T + ((size_t)nt0 * 16) * C_DIM + kt * 32, C_DIM);
    v16h B1 = load_bT_f16(w1T + ((size_t)(nt0 + 1) * 16) * C_DIM + kt * 32, C_DIM);
    a00 = wmma_f16(A0, B0, a00); a01 = wmma_f16(A0, B1, a01);
    a10 = wmma_f16(A1, B0, a10); a11 = wmma_f16(A1, B1, a11);
  }
  const int col = lane & 15, rbase = (lane < 16) ? 0 : 8;
#pragma unroll
  for (int mh = 0; mh < 2; ++mh) {
    const int mt = mh ? mt1 : mt0;
#pragma unroll
    for (int nh = 0; nh < 2; ++nh) {
      const v8f acc = mh ? (nh ? a11 : a10) : (nh ? a01 : a00);
      const int c = (nt0 + nh) * 16 + col;
      const float bias = b1[c];
#pragma unroll
      for (int j = 0; j < 8; ++j) {
        const int row = mt * 16 + rbase + j;
        const float t = acc[j] + bias;
        const float g = 0.5f * t * (1.f + erff(t * 0.70710678118654752f));
        h1[(size_t)row * 384 + c] = (_Float16)g;
      }
    }
  }
}

// ---------------- fc2 GEMM + residual -> d_out -------------------------------
// grid: (290, 3); 2x2 micro-tile; K = 384 = 12 x 32

__global__ void k_fc2(const _Float16* __restrict__ h1, const _Float16* __restrict__ w2T,
                      const float* __restrict__ b2, const float* __restrict__ xr,
                      float* __restrict__ out) {
  const int wave = threadIdx.x >> 5, lane = threadIdx.x & 31;
  const int mtp = blockIdx.x * 8 + wave;
  if (mtp >= (TP / 16) / 2) return;
  const int mt0 = mtp * 2, mt1 = mt0 + 1;
  const int nt0 = blockIdx.y * 2;
  __builtin_prefetch(w2T + (size_t)nt0 * 16 * 384, 0, 1);
  v8f a00 = {}, a01 = {}, a10 = {}, a11 = {};
#pragma unroll
  for (int kt = 0; kt < 12; ++kt) {
    v16h A0 = load_a_f16(h1 + (size_t)mt0 * 16 * 384 + kt * 32, 384);
    v16h A1 = load_a_f16(h1 + (size_t)mt1 * 16 * 384 + kt * 32, 384);
    v16h B0 = load_bT_f16(w2T + ((size_t)nt0 * 16) * 384 + kt * 32, 384);
    v16h B1 = load_bT_f16(w2T + ((size_t)(nt0 + 1) * 16) * 384 + kt * 32, 384);
    a00 = wmma_f16(A0, B0, a00); a01 = wmma_f16(A0, B1, a01);
    a10 = wmma_f16(A1, B0, a10); a11 = wmma_f16(A1, B1, a11);
  }
  const int col = lane & 15, rbase = (lane < 16) ? 0 : 8;
#pragma unroll
  for (int mh = 0; mh < 2; ++mh) {
    const int mt = mh ? mt1 : mt0;
#pragma unroll
    for (int nh = 0; nh < 2; ++nh) {
      const v8f acc = mh ? (nh ? a11 : a10) : (nh ? a01 : a00);
      const int c = (nt0 + nh) * 16 + col;
      const float bias = b2[c];
#pragma unroll
      for (int j = 0; j < 8; ++j) {
        const int row = mt * 16 + rbase + j;
        if (row < T_TOK)
          out[(size_t)row * C_DIM + c] = xr[(size_t)row * C_DIM + c] + acc[j] + bias;
      }
    }
  }
}

// ---------------- host launcher ----------------------------------------------

extern "C" void kernel_launch(void* const* d_in, const int* in_sizes, int n_in,
                              void* d_out, int out_size, void* d_ws, size_t ws_size,
                              hipStream_t stream) {
  (void)in_sizes; (void)n_in; (void)out_size; (void)ws_size;
  const float* x     = (const float*)d_in[0];
  const float* maskm = (const float*)d_in[1];
  const float* n1w   = (const float*)d_in[2];
  const float* n1b   = (const float*)d_in[3];
  const float* qkvw  = (const float*)d_in[4];
  const float* qkvb  = (const float*)d_in[5];
  const float* projw = (const float*)d_in[6];
  const float* projb = (const float*)d_in[7];
  const float* rpb   = (const float*)d_in[8];
  const float* n2w   = (const float*)d_in[9];
  const float* n2b   = (const float*)d_in[10];
  const float* f1w   = (const float*)d_in[11];
  const float* f1b   = (const float*)d_in[12];
  const float* f2w   = (const float*)d_in[13];
  const float* f2b   = (const float*)d_in[14];
  float* out = (float*)d_out;

  char* ws = (char*)d_ws;
  size_t off = 0;
  auto alloc = [&](size_t bytes) -> void* {
    void* p = ws + off;
    off = (off + bytes + 255) & ~(size_t)255;
    return p;
  };
  _Float16* qkvwT = (_Float16*)alloc((size_t)96 * 288 * 2);   // [288][96]
  _Float16* projwT = (_Float16*)alloc((size_t)96 * 96 * 2);   // [96][96]
  _Float16* f1wT = (_Float16*)alloc((size_t)96 * 384 * 2);    // [384][96]
  _Float16* f2wT = (_Float16*)alloc((size_t)384 * 96 * 2);    // [96][384]
  _Float16* hw = (_Float16*)alloc((size_t)TW * C_DIM * 2);    // LN1 windowed
  _Float16* ao = hw;                                          // alias: attn out
  float* xres = (float*)alloc((size_t)T_TOK * C_DIM * 4);
  _Float16* m16 = (_Float16*)alloc((size_t)TP * C_DIM * 2);
  size_t big = (size_t)3 * QS * 2;
  size_t h1b = (size_t)TP * 384 * 2;
  _Float16* qkv = (_Float16*)alloc(big > h1b ? big : h1b);    // qkv / h1 union
  _Float16* h1 = qkv;

  // weights -> f16, transposed to [N][K]
  k_cvt_t<<<dim3((288 + 255) / 256, 96), 256, 0, stream>>>(qkvw, qkvwT, 96, 288);
  k_cvt_t<<<dim3(1, 96), 256, 0, stream>>>(projw, projwT, 96, 96);
  k_cvt_t<<<dim3((384 + 255) / 256, 96), 256, 0, stream>>>(f1w, f1wT, 96, 384);
  k_cvt_t<<<dim3(1, 384), 256, 0, stream>>>(f2w, f2wT, 384, 96);

  // LN1 + shift + window: (44 n-blocks, 216 windows)
  k_ln1_window<<<dim3(NPAD / 8, NWIN), 256, 0, stream>>>(x, n1w, n1b, hw);
  // QKV: (4752/2/8, 18/2)
  k_qkv<<<dim3((TW / 16) / 2 / 8, 9), 256, 0, stream>>>(hw, qkvwT, qkvb, qkv);
  // attention: (648 win*head, 3 qblocks), 176 KB LDS
  k_attn<<<dim3(NWIN * NHEAD, 3), 256, 8 * 16 * NPAD * sizeof(float), stream>>>(
      qkv, rpb, maskm, ao);
  // proj + reverse + residual: (297, 3)
  k_proj<<<dim3((TW / 16) / 2 / 8, 3), 256, 0, stream>>>(ao, projwT, projb, x, xres);
  // LN2
  k_ln2<<<TP / 8, 256, 0, stream>>>(xres, n2w, n2b, m16);
  // fc1 + GELU: (290, 12)
  k_fc1<<<dim3(((TP / 16) / 2 + 7) / 8, 12), 256, 0, stream>>>(m16, f1wT, f1b, h1);
  // fc2 + residual -> out: (290, 3)
  k_fc2<<<dim3(((TP / 16) / 2 + 7) / 8, 3), 256, 0, stream>>>(h1, f2wT, f2b, xres, out);
}
